// DecoderWithAttention_57191784513931
// MI455X (gfx1250) — compile-verified, matrix operand order checked by reading, not run on hardware
//
#include <hip/hip_runtime.h>
#include <hip/hip_bf16.h>

// ---------------- problem constants ----------------
#define BB 64
#define PP 196          // 14*14
#define EE 2048
#define DD 512
#define AA 512
#define MM 512
#define VV 10000
#define SS 52
#define TT 51           // S-1

typedef __attribute__((ext_vector_type(16))) __bf16 v16bf;
typedef __attribute__((ext_vector_type(8)))  __bf16 v8bf;
typedef __attribute__((ext_vector_type(8)))  float  v8f;

__device__ __forceinline__ __bf16 f2bf(float f) {
    unsigned int u = __builtin_bit_cast(unsigned int, f);
    unsigned int lsb = (u >> 16) & 1u;
    u += 0x7fffu + lsb;                      // round-to-nearest-even
    unsigned short h = (unsigned short)(u >> 16);
    return __builtin_bit_cast(__bf16, h);
}

__device__ __forceinline__ float sigm(float x) { return 1.f / (1.f + __expf(-x)); }

// ---------------- kernel: stable descending sort of caption lengths ----------------
__global__ void sort_and_gather(const int* __restrict__ cap_len,
                                const int* __restrict__ caps,
                                int* __restrict__ sortind, int* __restrict__ declen_i,
                                int* __restrict__ caps_s,
                                float* __restrict__ out_caps, float* __restrict__ out_declen,
                                float* __restrict__ out_sortind) {
    __shared__ int s_ind[BB];
    int b = threadIdx.x;
    int len = cap_len[b];
    int rank = 0;
    for (int j = 0; j < BB; ++j) {
        int lj = cap_len[j];
        if (lj > len || (lj == len && j < b)) ++rank;   // stable argsort(-len)
    }
    s_ind[rank] = b;
    __syncthreads();
    int src = s_ind[b];
    sortind[b] = src;
    int dl = cap_len[src] - 1;
    declen_i[b] = dl;
    out_declen[b]  = (float)dl;
    out_sortind[b] = (float)src;
    for (int s = 0; s < SS; ++s) {
        int cv = caps[src * SS + s];
        caps_s[b * SS + s]   = cv;
        out_caps[b * SS + s] = (float)cv;
    }
}

// ---------------- kernel: row indirection map for the enc_att GEMM ----------------
__global__ void build_rowmap(const int* __restrict__ sortind, int* __restrict__ rowmap) {
    int idx = blockIdx.x * blockDim.x + threadIdx.x;
    if (idx >= BB * PP) return;
    int b = idx / PP, p = idx % PP;
    rowmap[idx] = sortind[b] * PP + p;
}

// ---------------- kernel: f32 -> bf16 conversion (vectorized) ----------------
__global__ void cvt_bf16_v4(const float* __restrict__ src, __bf16* __restrict__ dst, int n4) {
    int i = blockIdx.x * blockDim.x + threadIdx.x;   // one float4 per thread
    if (i >= n4) return;
    float4 v = ((const float4*)src)[i];
    __bf16 r0 = f2bf(v.x), r1 = f2bf(v.y), r2 = f2bf(v.z), r3 = f2bf(v.w);
    unsigned long long pk =
        (unsigned long long)__builtin_bit_cast(unsigned short, r0)
      | ((unsigned long long)__builtin_bit_cast(unsigned short, r1) << 16)
      | ((unsigned long long)__builtin_bit_cast(unsigned short, r2) << 32)
      | ((unsigned long long)__builtin_bit_cast(unsigned short, r3) << 48);
    ((unsigned long long*)dst)[i] = pk;
}

// ---------------- kernel: mean over pixels (with sort gather) -> bf16 ----------------
__global__ void mean_enc_k(const float* __restrict__ enc, const int* __restrict__ sortind,
                           __bf16* __restrict__ mean_bf) {
    int b = blockIdx.x;
    size_t base = (size_t)sortind[b] * PP * EE;
    for (int e = threadIdx.x; e < EE; e += blockDim.x) {
        float s = 0.f;
        for (int p = 0; p < PP; ++p) s += enc[base + (size_t)p * EE + e];
        mean_bf[b * EE + e] = f2bf(s * (1.f / (float)PP));
    }
}

// ---------------- generic bf16 WMMA GEMM: C[M,N] = A[M,K] @ W[N,K]^T (+bias)(+=C)(mask) ----
// A and W both bf16 row-major; one wave -> one 16x16 tile; k-loop of 32, unrolled x4.
__global__ __launch_bounds__(256) void gemm_bf16_wmma(
    const __bf16* __restrict__ A, const int* __restrict__ rowmap, int lda,
    const __bf16* __restrict__ Bw, int ldb,
    const float* __restrict__ bias,
    float* __restrict__ C, int ldc, __bf16* __restrict__ Cbf,
    int M, int N, int K,
    int accumulate, const int* __restrict__ declen, int tstep) {
    const int nTilesN = N >> 4;
    const int tiles   = (M >> 4) * nTilesN;
    const int waveId  = blockIdx.x * (blockDim.x >> 5) + (threadIdx.x >> 5);
    if (waveId >= tiles) return;                    // wave-uniform: EXEC stays all-1
    const int mT = waveId / nTilesN, nT = waveId % nTilesN;
    const int lane = threadIdx.x & 31;
    const int r16  = lane & 15;
    const bool hi  = lane >= 16;
    const int mBase = mT << 4, nBase = nT << 4;

    const int arow = rowmap ? rowmap[mBase + r16] : (mBase + r16);
    // ISA 16-bit fragment layout: lanes<16 hold K {0..7,16..23}, lanes>=16 hold {8..15,24..31}
    const __bf16* aRow = A  + (size_t)arow * lda          + (hi ? 8 : 0);
    const __bf16* bRow = Bw + (size_t)(nBase + r16) * ldb + (hi ? 8 : 0);

    v8f acc = {};
#pragma unroll 4
    for (int k = 0; k < K; k += 32) {
        const v8bf a0 = *(const v8bf*)(aRow + k);
        const v8bf a1 = *(const v8bf*)(aRow + k + 16);
        const v8bf b0 = *(const v8bf*)(bRow + k);
        const v8bf b1 = *(const v8bf*)(bRow + k + 16);
        __builtin_prefetch(aRow + k + 128, 0, 0);
        __builtin_prefetch(bRow + k + 128, 0, 0);
        v16bf av = __builtin_shufflevector(a0, a1, 0,1,2,3,4,5,6,7,8,9,10,11,12,13,14,15);
        v16bf bv = __builtin_shufflevector(b0, b1, 0,1,2,3,4,5,6,7,8,9,10,11,12,13,14,15);
        acc = __builtin_amdgcn_wmma_f32_16x16x32_bf16(
            /*neg_a=*/false, av, /*neg_b=*/false, bv,
            /*c_mod=*/(short)0, acc, /*reuse_a=*/false, /*reuse_b=*/false);
    }

    const int   ncol = nBase + r16;
    const float bcol = bias ? bias[ncol] : 0.f;
    const int   m0   = mBase + (hi ? 8 : 0);
#pragma unroll
    for (int r = 0; r < 8; ++r) {
        const int m = m0 + r;
        size_t off = (size_t)m * (size_t)ldc + (size_t)ncol;
        float v = acc[r] + bcol;
        if (accumulate) v += C[off];
        if (declen && tstep >= declen[m]) v = 0.f;  // mask inactive batch rows (preds)
        C[off] = v;
        if (Cbf) Cbf[off] = f2bf(v);
    }
}

// ---------------- kernel: attention scores + softmax + context z (one block per batch) ----
__global__ __launch_bounds__(256) void attention_step(
    const float* __restrict__ enc, const int* __restrict__ sortind,
    const float* __restrict__ enc_att, const float* __restrict__ dec_att,
    const float* __restrict__ Wf, const float* __restrict__ bf,
    const int* __restrict__ declen,
    float* __restrict__ z, float* __restrict__ alphas_out, int t) {
    __shared__ float dsh[AA];
    __shared__ float esh[PP];
    __shared__ float red[8];
    const int b = blockIdx.x;
    const int tid = threadIdx.x, lane = tid & 31, wave = tid >> 5;

    for (int a = tid; a < AA; a += 256) dsh[a] = dec_att[b * AA + a];
    __syncthreads();

    // e_p = tanh(enc_att + dec_att) . Wf + bf
    for (int p = wave; p < PP; p += 8) {
        const float* row = enc_att + ((size_t)b * PP + p) * AA;
        float accv = 0.f;
        for (int a = lane; a < AA; a += 32)
            accv += tanhf(row[a] + dsh[a]) * Wf[a];
        for (int m = 16; m >= 1; m >>= 1) accv += __shfl_xor(accv, m, 32);
        if (lane == 0) esh[p] = accv + bf[0];
    }
    __syncthreads();

    // softmax over 196
    float vmax = (tid < PP) ? esh[tid] : -3.4e38f;
    for (int m = 16; m >= 1; m >>= 1) vmax = fmaxf(vmax, __shfl_xor(vmax, m, 32));
    if (lane == 0) red[wave] = vmax;
    __syncthreads();
    float mx = red[0];
#pragma unroll
    for (int w = 1; w < 8; ++w) mx = fmaxf(mx, red[w]);
    __syncthreads();
    float ex = (tid < PP) ? __expf(esh[tid] - mx) : 0.f;
    float s = ex;
    for (int m = 16; m >= 1; m >>= 1) s += __shfl_xor(s, m, 32);
    if (lane == 0) red[wave] = s;
    __syncthreads();
    float tot = 0.f;
#pragma unroll
    for (int w = 0; w < 8; ++w) tot += red[w];
    float al = ex / tot;
    bool active = t < declen[b];
    if (tid < PP) {
        esh[tid] = al;                                  // reuse as alpha
        alphas_out[((size_t)b * TT + t) * PP + tid] = active ? al : 0.f;
    }
    __syncthreads();

    // z = sum_p alpha_p * enc[b_sorted, p, :]
    size_t base = (size_t)sortind[b] * PP * EE;
    for (int e = tid; e < EE; e += 256) {
        float accv = 0.f;
        for (int p = 0; p < PP; ++p) accv += esh[p] * enc[base + (size_t)p * EE + e];
        z[b * EE + e] = accv;
    }
}

// ---------------- kernel: x_bf = concat(emb_t, z) in bf16 ----------------
__global__ void build_x(const float* __restrict__ emb, const int* __restrict__ caps_s,
                        const float* __restrict__ z, __bf16* __restrict__ x_bf, int t) {
    int idx = blockIdx.x * blockDim.x + threadIdx.x;
    if (idx >= BB * (MM + EE)) return;
    int b = idx / (MM + EE), j = idx % (MM + EE);
    float v;
    if (j < MM) v = emb[(size_t)caps_s[b * SS + t] * MM + j];
    else        v = z[b * EE + (j - MM)];
    x_bf[idx] = f2bf(v);
}

// ---------------- kernel: LSTM cell + masked state update (dual f32/bf16 state) --------
__global__ void lstm_cell(const float* __restrict__ gates, float* __restrict__ h,
                          float* __restrict__ c, __bf16* __restrict__ h_bf,
                          __bf16* __restrict__ hnew_bf,
                          const int* __restrict__ declen, int t) {
    int idx = blockIdx.x * blockDim.x + threadIdx.x;
    if (idx >= BB * DD) return;
    int b = idx >> 9;          // /512
    int j = idx & (DD - 1);
    const float* g = gates + (size_t)b * 4 * DD;
    float ig = sigm(g[j]);
    float fg = sigm(g[j + DD]);
    float gg = tanhf(g[j + 2 * DD]);
    float og = sigm(g[j + 3 * DD]);
    float cn = fg * c[idx] + ig * gg;
    float hn = og * tanhf(cn);
    hnew_bf[idx] = f2bf(hn);
    bool active = t < declen[b];
    float hout = active ? hn : h[idx];
    float cout = active ? cn : c[idx];
    h[idx] = hout;
    c[idx] = cout;
    h_bf[idx] = f2bf(hout);
}

// =====================================================================
extern "C" void kernel_launch(void* const* d_in, const int* in_sizes, int n_in,
                              void* d_out, int out_size, void* d_ws, size_t ws_size,
                              hipStream_t stream) {
    (void)in_sizes; (void)n_in; (void)out_size; (void)ws_size;
    // ---- inputs (setup_inputs order) ----
    const float* enc        = (const float*)d_in[0];
    const int*   caps       = (const int*)  d_in[1];
    const int*   cap_len    = (const int*)  d_in[2];
    const float* emb_table  = (const float*)d_in[3];
    const float* W_enc_att  = (const float*)d_in[4];
    const float* b_enc_att  = (const float*)d_in[5];
    const float* W_dec_att  = (const float*)d_in[6];
    const float* b_dec_att  = (const float*)d_in[7];
    const float* W_f_att    = (const float*)d_in[8];
    const float* b_f_att    = (const float*)d_in[9];
    const float* W_ih       = (const float*)d_in[10];
    const float* b_ih       = (const float*)d_in[11];
    const float* W_hh       = (const float*)d_in[12];
    const float* b_hh       = (const float*)d_in[13];
    const float* W_init_h   = (const float*)d_in[14];
    const float* b_init_h   = (const float*)d_in[15];
    const float* W_init_c   = (const float*)d_in[16];
    const float* b_init_c   = (const float*)d_in[17];
    const float* W_fc       = (const float*)d_in[18];
    const float* b_fc       = (const float*)d_in[19];

    // ---- output layout (concatenated, float) ----
    float* out = (float*)d_out;
    float* out_pred   = out;                                        // B*T*V
    float* out_caps   = out_pred   + (size_t)BB * TT * VV;          // B*S
    float* out_declen = out_caps   + (size_t)BB * SS;               // B
    float* out_alphas = out_declen + (size_t)BB;                    // B*T*P
    float* out_sort   = out_alphas + (size_t)BB * TT * PP;          // B

    // ---- workspace carve-up ----
    char* ws = (char*)d_ws;
    size_t o = 0;
    auto carve = [&](size_t bytes) { char* p = ws + o; o = (o + bytes + 255) & ~(size_t)255; return p; };
    __bf16* enc_bf  = (__bf16*)carve((size_t)BB * PP * EE * 2);     // 51 MB, L2-resident
    __bf16* Wenc_bf = (__bf16*)carve((size_t)AA * EE * 2);
    __bf16* Wdec_bf = (__bf16*)carve((size_t)AA * DD * 2);
    __bf16* Wih_bf  = (__bf16*)carve((size_t)4 * DD * (MM + EE) * 2);
    __bf16* Whh_bf  = (__bf16*)carve((size_t)4 * DD * DD * 2);
    __bf16* Wfc_bf  = (__bf16*)carve((size_t)VV * DD * 2);
    __bf16* Winh_bf = (__bf16*)carve((size_t)DD * EE * 2);
    __bf16* Winc_bf = (__bf16*)carve((size_t)DD * EE * 2);
    __bf16* mean_bf = (__bf16*)carve((size_t)BB * EE * 2);
    __bf16* h_bf    = (__bf16*)carve((size_t)BB * DD * 2);
    __bf16* hnew_bf = (__bf16*)carve((size_t)BB * DD * 2);
    __bf16* x_bf    = (__bf16*)carve((size_t)BB * (MM + EE) * 2);
    float* enc_att  = (float*)carve((size_t)BB * PP * AA * 4);
    float* h        = (float*)carve((size_t)BB * DD * 4);
    float* c        = (float*)carve((size_t)BB * DD * 4);
    float* dec_att  = (float*)carve((size_t)BB * AA * 4);
    float* z        = (float*)carve((size_t)BB * EE * 4);
    float* gates    = (float*)carve((size_t)BB * 4 * DD * 4);
    int* sortind    = (int*)carve(BB * 4);
    int* declen_i   = (int*)carve(BB * 4);
    int* caps_s     = (int*)carve(BB * SS * 4);
    int* rowmap     = (int*)carve(BB * PP * 4);

    auto gemm = [&](const __bf16* A, const int* rmap, int lda,
                    const __bf16* Bw, int ldb, const float* bias,
                    float* C, int ldc, __bf16* Cbf, int M, int N, int K,
                    int accum, const int* mask, int t) {
        int tiles  = (M / 16) * (N / 16);
        int blocks = (tiles + 7) / 8;   // 8 waves (256 thr) per block
        gemm_bf16_wmma<<<blocks, 256, 0, stream>>>(A, rmap, lda, Bw, ldb, bias,
                                                   C, ldc, Cbf, M, N, K, accum, mask, t);
    };
    auto cvt = [&](const float* src, __bf16* dst, size_t n) {
        size_t n4 = n / 4;   // all our sizes are multiples of 4
        cvt_bf16_v4<<<(int)((n4 + 255) / 256), 256, 0, stream>>>(src, dst, (int)n4);
    };

    // ---- prologue ----
    sort_and_gather<<<1, BB, 0, stream>>>(cap_len, caps, sortind, declen_i, caps_s,
                                          out_caps, out_declen, out_sort);
    build_rowmap<<<(BB * PP + 255) / 256, 256, 0, stream>>>(sortind, rowmap);
    cvt(enc,       enc_bf,  (size_t)BB * PP * EE);
    cvt(W_enc_att, Wenc_bf, (size_t)AA * EE);
    cvt(W_dec_att, Wdec_bf, (size_t)AA * DD);
    cvt(W_ih,      Wih_bf,  (size_t)4 * DD * (MM + EE));
    cvt(W_hh,      Whh_bf,  (size_t)4 * DD * DD);
    cvt(W_fc,      Wfc_bf,  (size_t)VV * DD);
    cvt(W_init_h,  Winh_bf, (size_t)DD * EE);
    cvt(W_init_c,  Winc_bf, (size_t)DD * EE);
    mean_enc_k<<<BB, 256, 0, stream>>>(enc, sortind, mean_bf);

    // h0 / c0  (h0 epilogue also produces bf16 h for the first dec_att / W_hh GEMMs)
    gemm(mean_bf, nullptr, EE, Winh_bf, EE, b_init_h, h, DD, h_bf, BB, DD, EE, 0, nullptr, 0);
    gemm(mean_bf, nullptr, EE, Winc_bf, EE, b_init_c, c, DD, nullptr, BB, DD, EE, 0, nullptr, 0);
    // enc_att = enc_sorted @ W_enc_att^T  (sort folded in via rowmap)
    gemm(enc_bf, rowmap, EE, Wenc_bf, EE, b_enc_att, enc_att, AA, nullptr,
         BB * PP, AA, EE, 0, nullptr, 0);

    // ---- time loop ----
    for (int t = 0; t < TT; ++t) {
        gemm(h_bf, nullptr, DD, Wdec_bf, DD, b_dec_att, dec_att, AA, nullptr,
             BB, AA, DD, 0, nullptr, 0);
        attention_step<<<BB, 256, 0, stream>>>(enc, sortind, enc_att, dec_att,
                                               W_f_att, b_f_att, declen_i,
                                               z, out_alphas, t);
        build_x<<<(BB * (MM + EE) + 255) / 256, 256, 0, stream>>>(emb_table, caps_s, z, x_bf, t);
        gemm(x_bf, nullptr, MM + EE, Wih_bf, MM + EE, b_ih, gates, 4 * DD, nullptr,
             BB, 4 * DD, MM + EE, 0, nullptr, 0);
        gemm(h_bf, nullptr, DD, Whh_bf, DD, b_hh, gates, 4 * DD, nullptr,
             BB, 4 * DD, DD, 1, nullptr, 0);
        lstm_cell<<<(BB * DD + 255) / 256, 256, 0, stream>>>(gates, h, c, h_bf, hnew_bf,
                                                             declen_i, t);
        // preds written masked directly into d_out: row b at out_pred + b*T*V + t*V
        gemm(hnew_bf, nullptr, DD, Wfc_bf, DD, b_fc, out_pred + (size_t)t * VV, TT * VV,
             nullptr, BB, VV, DD, 0, declen_i, t);
    }
}
